// SuperWarpF_5480378269848
// MI455X (gfx1250) — matrix operand order repaired and checked by reading, not compiled, not run on hardware
//
#include <hip/hip_runtime.h>

// ---------------------------------------------------------------------------
// SuperWarpF for MI455X (gfx1250): implicit-GEMM masked 3x3 convs on
// v_wmma_f32_16x16x32_f16, wave32, with GLOBAL_LOAD_ASYNC_TO_LDS staging of
// f16 NHWC activations and tap-major packed f16 weights.
// ---------------------------------------------------------------------------

typedef __attribute__((ext_vector_type(16))) _Float16 v16h;
typedef __attribute__((ext_vector_type(8)))  float    v8f;

constexpr int KB  = 2;      // batch
constexpr int KC  = 64;     // channels per pyramid
constexpr int KH  = 256;
constexpr int KW  = 256;
constexpr int KHW = KH * KW;
constexpr int KP  = 3;      // pyramids
constexpr int KNO = 16;     // per-pyramid output channels
constexpr int KFEX = KP * KNO;  // 48
constexpr int KN  = 32768;  // sparse points

// --- gfx1250 async global->LDS copy (ASYNCcnt path) ------------------------
__device__ __forceinline__ void async_ld_b128_to_lds(unsigned lds_off,
                                                     unsigned long long gaddr) {
  asm volatile("global_load_async_to_lds_b128 %0, %1, off"
               :: "v"(lds_off), "v"(gaddr) : "memory");
}
__device__ __forceinline__ void wait_async_zero() {
  asm volatile("s_wait_asynccnt 0x0" ::: "memory");
}

// ---------------------------------------------------------------------------
// wm = mask / clamp(avg3x3(mask), 1e-8)
// ---------------------------------------------------------------------------
__global__ void wm_kernel(const float* __restrict__ mask, float* __restrict__ wm) {
  int pix = blockIdx.x * blockDim.x + threadIdx.x;
  if (pix >= KHW) return;
  int y = pix >> 8, x = pix & 255;
  float s = 0.f;
  #pragma unroll
  for (int dy = -1; dy <= 1; ++dy)
    #pragma unroll
    for (int dx = -1; dx <= 1; ++dx) {
      int py = y + dy, px = x + dx;
      if (py >= 0 && py < KH && px >= 0 && px < KW) s += mask[py * KW + px];
    }
  s *= (1.f / 9.f);
  wm[pix] = mask[pix] / fmaxf(s, 1e-8f);
}

// ---------------------------------------------------------------------------
// Pack conv weights (COUT,CIN,3,3) f32 -> tap-major padded f16 (9,COUT,64)
// ---------------------------------------------------------------------------
__global__ void pack_w_kernel(const float* __restrict__ src, _Float16* __restrict__ dst,
                              int cout, int cin) {
  int i = blockIdx.x * blockDim.x + threadIdx.x;
  int tot = 9 * cout * 64;
  if (i >= tot) return;
  int tap = i / (cout * 64);
  int rem = i - tap * cout * 64;
  int co = rem >> 6, ci = rem & 63;
  float v = (ci < cin) ? src[(co * cin + ci) * 9 + tap] : 0.f;
  dst[i] = (_Float16)v;
}

// ---------------------------------------------------------------------------
// Pack masked pyramid input: (B,64,HW) f32 * mask -> (B,HW,64) f16 NHWC
// ---------------------------------------------------------------------------
__global__ void pack_x_kernel(const float* __restrict__ x, const float* __restrict__ mask,
                              _Float16* __restrict__ xh) {
  int i = blockIdx.x * blockDim.x + threadIdx.x;
  if (i >= KB * KHW * 64) return;
  int c   = i & 63;
  int pix = (i >> 6) % KHW;
  int b   = i / (KHW * 64);
  xh[i] = (_Float16)(x[((size_t)b * 64 + c) * KHW + pix] * mask[pix]);
}

// ---------------------------------------------------------------------------
// Masked 3x3 conv as implicit GEMM on WMMA f16 (f32 accumulate).
// A: f16 NHWC activations (B,HW,64), zero-padded channels; staged per tap
//    into LDS with async b128 copies (borders zero-filled via DS stores).
// B: packed f16 weights (9,COUT,64); contiguous tap tile, async b128 copies.
//   out_f32[b, cofs+co, pix] = conv(x)[co,pix] * wm[pix] (+relu) (+resid)
//   out_h  [b, pix, houtofs+co] = f16(val)   (NHWC copy for the next conv)
// Block: 256 threads = 8 wave32; wave = one 16-pixel M-tile; K = 9 taps x 64.
// ---------------------------------------------------------------------------
template <int COUT>
__global__ __launch_bounds__(256)
void masked_conv3x3_wmma(const _Float16* __restrict__ xh,   // (B,HW,64)
                         const _Float16* __restrict__ wp,   // (9,COUT,64)
                         const float* __restrict__ mask,
                         const float* __restrict__ wm,
                         const float* __restrict__ resid,   // nullable (B,COUT,HW)
                         float* __restrict__ outf,          // (B,cbuf,HW)
                         _Float16* __restrict__ outh,       // nullable (B,HW,64)
                         int cbuf, int cofs, int houtofs,
                         int premask_resid, int do_relu, int pad_hi) {
  constexpr int NT = COUT / 16;

  __shared__ alignas(16) _Float16 As[8][16][64];   // 16 KB
  __shared__ alignas(16) _Float16 Bs[COUT][64];    // <= 8 KB

  const int tid  = threadIdx.x;
  const int wave = tid >> 5;
  const int lane = tid & 31;
  const int wg   = blockIdx.x;                  // B*HW/128 blocks
  const int b    = wg / (KHW / 128);
  const int pixbase = (wg % (KHW / 128)) * 128;
  const int wpix = pixbase + wave * 16;

  const _Float16* xb = xh + (size_t)b * KHW * 64;

  v8f acc[NT];
  const v8f vzero = {0.f, 0.f, 0.f, 0.f, 0.f, 0.f, 0.f, 0.f};
  #pragma unroll
  for (int t = 0; t < NT; ++t) acc[t] = vzero;

  const int row  = lane & 15;
  const int half = lane >> 4;

  for (int tap = 0; tap < 9; ++tap) {
    const int dy = tap / 3 - 1, dx = tap % 3 - 1;

    // --- async-stage tap weight tile (contiguous, COUT*8 x 16B chunks) -----
    for (int cch = tid; cch < COUT * 8; cch += 256) {
      unsigned lds = (unsigned)(uintptr_t)&Bs[0][0] + (unsigned)cch * 16u;
      unsigned long long g =
          (unsigned long long)(uintptr_t)(wp + ((size_t)tap * COUT * 64 + cch * 8));
      async_ld_b128_to_lds(lds, g);
    }

    // --- async-stage this wave's 16 activation rows (128B per pixel) -------
    #pragma unroll
    for (int i = 0; i < 4; ++i) {
      const int chunk = lane + 32 * i;          // 0..127
      const int p     = chunk >> 3;             // pixel 0..15
      const int part  = chunk & 7;              // 16B part within 128B row
      const int pix   = wpix + p;
      const int py    = (pix >> 8) + dy;
      const int px    = (pix & 255) + dx;
      if (py >= 0 && py < KH && px >= 0 && px < KW) {
        unsigned lds = (unsigned)(uintptr_t)&As[wave][p][part * 8];
        unsigned long long g = (unsigned long long)(uintptr_t)(
            xb + ((size_t)(py * KW + px) * 64 + part * 8));
        async_ld_b128_to_lds(lds, g);
      } else {
        float4 z = make_float4(0.f, 0.f, 0.f, 0.f);
        *reinterpret_cast<float4*>(&As[wave][p][part * 8]) = z;   // border zero
      }
    }

    wait_async_zero();
    __syncthreads();

    // --- 2 k-steps of 32 over the 64 padded input channels -----------------
    #pragma unroll
    for (int kk = 0; kk < 2; ++kk) {
      v16h a;
      #pragma unroll
      for (int e = 0; e < 16; ++e) {   // ISA 16-bit A layout (16x32)
        const int k32 = (e < 8) ? (half * 8 + e) : (16 + half * 8 + (e - 8));
        a[e] = As[wave][row][kk * 32 + k32];
      }
      #pragma unroll
      for (int t = 0; t < NT; ++t) {
        v16h bf;
        const int n = t * 16 + row;
        #pragma unroll
        for (int e = 0; e < 16; ++e) { // ISA 16-bit B layout (32x16)
          const int k32 = (e < 8) ? (half * 8 + e) : (16 + half * 8 + (e - 8));
          bf[e] = Bs[n][kk * 32 + k32];
        }
        acc[t] = __builtin_amdgcn_wmma_f32_16x16x32_f16(
            false, a, false, bf, (short)0, acc[t], false, false);
      }
    }
    __syncthreads();
  }

  // --- epilogue: *wm, relu, residual, f32 + f16-NHWC stores ----------------
  #pragma unroll
  for (int t = 0; t < NT; ++t) {
    const int co = t * 16 + row;
    #pragma unroll
    for (int r = 0; r < 8; ++r) {
      const int p   = r + 8 * half;              // C/D layout: M = r + 8*half
      const int pix = wpix + p;
      float val = acc[t][r] * wm[pix];
      if (do_relu) val = fmaxf(val, 0.f);
      if (resid) {
        float rv = resid[((size_t)b * COUT + co) * KHW + pix];
        if (premask_resid) rv *= mask[pix];
        val += rv;
      }
      outf[((size_t)b * cbuf + cofs + co) * KHW + pix] = val;
      if (outh) outh[((size_t)b * KHW + pix) * 64 + houtofs + co] = (_Float16)val;
    }
  }
  if (outh && pad_hi) {          // zero the padded channels 48..63 (NHWC)
    const int co = COUT + row;
    if (co < 64) {
      #pragma unroll
      for (int r = 0; r < 8; ++r) {
        const int pix = wpix + r + 8 * half;
        outh[((size_t)b * KHW + pix) * 64 + co] = (_Float16)0.f;
      }
    }
  }
}

// ---------------------------------------------------------------------------
__global__ void add_kernel(const float* __restrict__ a, const float* __restrict__ b,
                           float* __restrict__ dst, int n) {
  int i = blockIdx.x * blockDim.x + threadIdx.x;
  if (i < n) dst[i] = a[i] + b[i];
}

__global__ void zero_kernel(float* __restrict__ p, int n) {
  int i = blockIdx.x * blockDim.x + threadIdx.x;
  if (i < n) p[i] = 0.f;
}

// ---------------------------------------------------------------------------
// MLP layer 1: gather 48 features at yi + dsda -> 49-in, 48-out, relu
// ---------------------------------------------------------------------------
__global__ void mlp1_kernel(const float* __restrict__ f,    // (B,48,HW)
                            const int* __restrict__ yi,
                            const float* __restrict__ m,    // 9
                            const float* __restrict__ w1,   // (48,49)
                            const float* __restrict__ b1,
                            float* __restrict__ h1) {       // (B,48,N)
  int i = blockIdx.x * blockDim.x + threadIdx.x;
  if (i >= KB * KN) return;
  int b = i / KN, n = i % KN;
  int j = yi[n];

  float in_[49];
  #pragma unroll
  for (int c = 0; c < KFEX; ++c) in_[c] = f[((size_t)b * KFEX + c) * KHW + j];

  float det = m[0] * (m[4] * m[8] - m[5] * m[7])
            - m[1] * (m[3] * m[8] - m[5] * m[6])
            + m[2] * (m[3] * m[7] - m[4] * m[6]);
  float yy = (float)(j >> 8), xx = (float)(j & 255);
  float wc = m[6] * xx + m[7] * yy + m[8];
  float aw = fmaxf(fabsf(wc), 1e-8f);
  float d  = fabsf(det) / (aw * aw * aw);
  in_[48] = logf(d + 1e-8f);

  for (int o = 0; o < KFEX; ++o) {
    float s = b1[o];
    #pragma unroll
    for (int c = 0; c < 49; ++c) s += w1[o * 49 + c] * in_[c];
    h1[((size_t)b * KFEX + o) * KN + n] = fmaxf(s, 0.f);
  }
}

__global__ void mlp2_kernel(const float* __restrict__ h1, const float* __restrict__ w2,
                            const float* __restrict__ b2, float* __restrict__ h2) {
  int i = blockIdx.x * blockDim.x + threadIdx.x;
  if (i >= KB * KN) return;
  int b = i / KN, n = i % KN;
  float in_[KFEX];
  #pragma unroll
  for (int c = 0; c < KFEX; ++c) in_[c] = h1[((size_t)b * KFEX + c) * KN + n];
  for (int o = 0; o < KFEX; ++o) {
    float s = b2[o];
    #pragma unroll
    for (int c = 0; c < KFEX; ++c) s += w2[o * KFEX + c] * in_[c];
    h2[((size_t)b * KFEX + o) * KN + n] = fmaxf(s, 0.f);
  }
}

__global__ void mlp3_scatter_kernel(const float* __restrict__ h2,
                                    const int* __restrict__ yi,
                                    const float* __restrict__ w3,
                                    const float* __restrict__ b3,
                                    float* __restrict__ wsd) {
  int i = blockIdx.x * blockDim.x + threadIdx.x;
  if (i >= KB * KN) return;
  int b = i / KN, n = i % KN;
  int j = yi[n];
  float in_[KFEX];
  #pragma unroll
  for (int c = 0; c < KFEX; ++c) in_[c] = h2[((size_t)b * KFEX + c) * KN + n];
  #pragma unroll
  for (int o = 0; o < KP; ++o) {
    float s = b3[o];
    #pragma unroll
    for (int c = 0; c < KFEX; ++c) s += w3[o * KFEX + c] * in_[c];
    wsd[((size_t)b * KP + o) * KHW + j] = s;
  }
}

// ---------------------------------------------------------------------------
// out[b,c,pix] = sum_p ws[b,p,pix] * xs[p,b,c,pix]
// ---------------------------------------------------------------------------
__global__ void blend_kernel(const float* __restrict__ wsd,
                             const float* __restrict__ xs,
                             float* __restrict__ out) {
  int i = blockIdx.x * blockDim.x + threadIdx.x;
  if (i >= KB * KC * KHW) return;
  int pix = i % KHW;
  int c   = (i / KHW) % KC;
  int b   = i / (KHW * KC);
  float s = 0.f;
  #pragma unroll
  for (int p = 0; p < KP; ++p) {
    s += wsd[((size_t)b * KP + p) * KHW + pix] *
         xs[(((size_t)p * KB + b) * KC + c) * KHW + pix];
  }
  out[i] = s;
}

// ---------------------------------------------------------------------------
// launch
// ---------------------------------------------------------------------------
extern "C" void kernel_launch(void* const* d_in, const int* in_sizes, int n_in,
                              void* d_out, int out_size, void* d_ws, size_t ws_size,
                              hipStream_t stream) {
  (void)in_sizes; (void)n_in; (void)out_size; (void)ws_size;

  const float* xs    = (const float*)d_in[0];   // (P,B,64,H,W)
  const float* mask  = (const float*)d_in[1];   // (1,1,H,W)
  const int*   yi    = (const int*)d_in[2];     // (N,)
  const float* m     = (const float*)d_in[3];   // (3,3)
  const float* ssres = (const float*)d_in[4];   // (P,2,2,64,64,3,3)
  const float* ssout = (const float*)d_in[5];   // (P,16,64,3,3)
  const float* fres  = (const float*)d_in[6];   // (2,2,48,48,3,3)
  const float* w1    = (const float*)d_in[7];
  const float* b1    = (const float*)d_in[8];
  const float* w2    = (const float*)d_in[9];
  const float* b2    = (const float*)d_in[10];
  const float* w3    = (const float*)d_in[11];
  const float* b3    = (const float*)d_in[12];
  float* out = (float*)d_out;

  // ---- scratch layout ----
  char* ws = (char*)d_ws;
  const size_t SZ64 = (size_t)KB * 64 * KHW * 4;      // 32 MB f32
  const size_t SZ48 = (size_t)KB * KFEX * KHW * 4;    // 24 MB f32
  const size_t SZH  = (size_t)KB * KHW * 64 * 2;      // 16 MB f16 NHWC
  float*    WM = (float*)(ws);
  float*    Rb = (float*)(ws + 262144);
  float*    Tb = (float*)(ws + 262144 + SZ64);
  float*    Ub = (float*)(ws + 262144 + 2 * SZ64);
  float*    Fb = (float*)(ws + 262144 + 3 * SZ64);
  _Float16* XH = (_Float16*)(ws + 262144 + 3 * SZ64 + SZ48);
  _Float16* RH = (_Float16*)((char*)XH + SZH);
  _Float16* TH = (_Float16*)((char*)RH + SZH);
  _Float16* FH = (_Float16*)((char*)TH + SZH);
  _Float16* WP = (_Float16*)((char*)FH + SZH);
  _Float16* UH = XH;   // reuse: c4's NHWC output (XH no longer needed then)

  // packed-weight sub-offsets (halfs)
  const size_t PW64 = (size_t)9 * 64 * 64;   // per 64->64 conv
  const size_t PW16 = (size_t)9 * 16 * 64;
  const size_t PW48 = (size_t)9 * 48 * 64;
  _Float16* wpres = WP;                        // 12 x PW64
  _Float16* wpout = WP + 12 * PW64;            // 3 x PW16
  _Float16* wpf   = WP + 12 * PW64 + 3 * PW16; // 4 x PW48

  const dim3 cgrid(KB * KHW / 128), cblk(256);

  wm_kernel<<<KHW / 256, 256, 0, stream>>>(mask, WM);

  // ---- pack all weights to tap-major padded f16 ----
  const size_t WRES = (size_t)64 * 64 * 9;
  for (int k = 0; k < 12; ++k)
    pack_w_kernel<<<(int)((PW64 + 255) / 256), 256, 0, stream>>>(
        ssres + (size_t)k * WRES, wpres + (size_t)k * PW64, 64, 64);
  for (int k = 0; k < 3; ++k)
    pack_w_kernel<<<(int)((PW16 + 255) / 256), 256, 0, stream>>>(
        ssout + (size_t)k * KNO * 64 * 9, wpout + (size_t)k * PW16, 16, 64);
  for (int k = 0; k < 4; ++k)
    pack_w_kernel<<<(int)((PW48 + 255) / 256), 256, 0, stream>>>(
        fres + (size_t)k * KFEX * KFEX * 9, wpf + (size_t)k * PW48, 48, 48);

  // zero FH once (c5 fills 16-channel slices; channels 48..63 stay zero)
  zero_kernel<<<(int)(SZH / 4 + 255) / 256, 256, 0, stream>>>((float*)FH, (int)(SZH / 4));

  // ---- pyramids ----
  for (int i = 0; i < KP; ++i) {
    const float* xi = xs + (size_t)i * KB * KC * KHW;
    _Float16* w00 = wpres + (size_t)(i * 4 + 0) * PW64;
    _Float16* w01 = wpres + (size_t)(i * 4 + 1) * PW64;
    _Float16* w10 = wpres + (size_t)(i * 4 + 2) * PW64;
    _Float16* w11 = wpres + (size_t)(i * 4 + 3) * PW64;
    _Float16* wo  = wpout + (size_t)i * PW16;

    pack_x_kernel<<<KB * KHW * 64 / 256, 256, 0, stream>>>(xi, mask, XH);
    // block 1
    masked_conv3x3_wmma<64><<<cgrid, cblk, 0, stream>>>(
        XH, w00, mask, WM, nullptr, Rb, RH, 64, 0, 0, 0, 1, 0);
    masked_conv3x3_wmma<64><<<cgrid, cblk, 0, stream>>>(
        RH, w01, mask, WM, xi, Tb, TH, 64, 0, 0, 1, 0, 0);
    // block 2
    masked_conv3x3_wmma<64><<<cgrid, cblk, 0, stream>>>(
        TH, w10, mask, WM, nullptr, Rb, RH, 64, 0, 0, 0, 1, 0);
    masked_conv3x3_wmma<64><<<cgrid, cblk, 0, stream>>>(
        RH, w11, mask, WM, Tb, Ub, UH, 64, 0, 0, 1, 0, 0);
    // 64 -> 16 projection into F slice
    masked_conv3x3_wmma<16><<<cgrid, cblk, 0, stream>>>(
        UH, wo, mask, WM, nullptr, Fb, FH, KFEX, KNO * i, KNO * i, 0, 0, 0);
  }

  // ---- shared 48-channel trunk ----
  masked_conv3x3_wmma<48><<<cgrid, cblk, 0, stream>>>(
      FH, wpf + 0 * PW48, mask, WM, nullptr, Rb, RH, KFEX, 0, 0, 0, 1, 1);
  masked_conv3x3_wmma<48><<<cgrid, cblk, 0, stream>>>(
      RH, wpf + 1 * PW48, mask, WM, Fb, Tb, TH, KFEX, 0, 0, 1, 0, 1);
  masked_conv3x3_wmma<48><<<cgrid, cblk, 0, stream>>>(
      TH, wpf + 2 * PW48, mask, WM, nullptr, Rb, RH, KFEX, 0, 0, 0, 1, 1);
  masked_conv3x3_wmma<48><<<cgrid, cblk, 0, stream>>>(
      RH, wpf + 3 * PW48, mask, WM, Tb, Ub, nullptr, KFEX, 0, 0, 1, 0, 0);

  // f_final = F + g -> Rb
  const int nadd = KB * KFEX * KHW;
  add_kernel<<<(nadd + 255) / 256, 256, 0, stream>>>(Fb, Ub, Rb, nadd);

  // ---- sampler MLP over sparse points ----
  const int npts = KB * KN;
  mlp1_kernel<<<(npts + 255) / 256, 256, 0, stream>>>(Rb, yi, m, w1, b1, Fb);
  mlp2_kernel<<<(npts + 255) / 256, 256, 0, stream>>>(Fb, w2, b2, Tb);
  const int nws = KB * KP * KHW;
  zero_kernel<<<(nws + 255) / 256, 256, 0, stream>>>(Ub, nws);
  mlp3_scatter_kernel<<<(npts + 255) / 256, 256, 0, stream>>>(Tb, yi, w3, b3, Ub);

  // ---- blend pyramids ----
  const int nout = KB * KC * KHW;
  blend_kernel<<<(nout + 255) / 256, 256, 0, stream>>>(Ub, xs, out);
}
